// DiffPoolBatchedGraphLayer_16192026705962
// MI455X (gfx1250) — matrix operand order repaired and checked by reading, not compiled, not run on hardware
//
#include <hip/hip_runtime.h>
#include <hip/hip_bf16.h>

typedef __attribute__((ext_vector_type(16))) _Float16 v16h;
typedef __attribute__((ext_vector_type(8)))  _Float16 v8h;
typedef __attribute__((ext_vector_type(8)))  float    v8f;

#define B_    64
#define N_    256
#define DIN_  256
#define K_    128
#define E_    8192
#define AK_   (2*DIN_)        // 512 (concat h|agg)
#define ROWS_ (B_*N_)         // 16384
#define ADJ_  ((long)B_*K_)   // 8192

#define USE_ASYNC_LDS 1       // inline-asm global_load_async_to_lds_b128 path

__device__ __forceinline__ float atomAddDev(float* p, float v) {
  return __hip_atomic_fetch_add(p, v, __ATOMIC_RELAXED, __HIP_MEMORY_SCOPE_AGENT);
}
__device__ __forceinline__ float atomAddLds(float* p, float v) {
  return __hip_atomic_fetch_add(p, v, __ATOMIC_RELAXED, __HIP_MEMORY_SCOPE_WORKGROUP);
}

// A-fragment (16x32 f16, row-major source, rows = M). Per ISA 7.12.2:
// lanes 0-15: row m=lane, K {0..7} then {16..23}; lanes 16-31: row m=lane-16, K {8..15} then {24..31}.
__device__ __forceinline__ v16h load_a_frag(const _Float16* base, int stride, int lane) {
  int m    = lane & 15;
  int koff = (lane & 16) ? 8 : 0;
  const _Float16* p = base + (long)m * stride + koff;
  v8h lo = *(const v8h*)(p);
  v8h hi = *(const v8h*)(p + 16);
  return __builtin_shufflevector(lo, hi, 0,1,2,3,4,5,6,7,8,9,10,11,12,13,14,15);
}

// B-fragment (32x16 f16). Source is B^T row-major (rows = N columns of B, stride in halfs).
// lanes 0-15: column n=lane, K {0..7} then {16..23}; lanes 16-31: K {8..15} then {24..31}.
__device__ __forceinline__ v16h load_b_frag(const _Float16* baseT, int stride, int lane) {
  int n    = lane & 15;
  int koff = (lane & 16) ? 8 : 0;
  const _Float16* p = baseT + (long)n * stride + koff;
  v8h lo = *(const v8h*)(p);
  v8h hi = *(const v8h*)(p + 16);
  return __builtin_shufflevector(lo, hi, 0,1,2,3,4,5,6,7,8,9,10,11,12,13,14,15);
}

__device__ __forceinline__ v8f wmma_f16(v16h a, v16h b, v8f c) {
  return __builtin_amdgcn_wmma_f32_16x16x32_f16(false, a, false, b, (short)0, c, false, false);
}

// ---------------- zero fill (268MB block-diagonal adjacency) ----------------
__global__ void zero4_kernel(float4* p, long n4) {
  long i = (long)blockIdx.x * blockDim.x + threadIdx.x;
  if (i < n4) p[i] = make_float4(0.f, 0.f, 0.f, 0.f);
}

// ---------------- fused scatter-mean + A build, one WG per graph ----------------
// 257KB dynamic LDS: agg[256][256] f32 + deg[256] f32. LDS float atomics instead of
// L2 atomics; emits Af = f16([h | agg/deg]) directly.
__global__ __launch_bounds__(256) void scatter_build_kernel(const float* __restrict__ h,
                                                            const int* __restrict__ esrc,
                                                            const int* __restrict__ edst,
                                                            _Float16* __restrict__ Af) {
  extern __shared__ float smem[];               // [N_*DIN_ agg][N_ deg]
  float* aggS = smem;
  float* degS = smem + N_ * DIN_;
  int b = blockIdx.x;
  int t = threadIdx.x, w = t >> 5, lane = t & 31;
  for (int i = t; i < N_ * DIN_ + N_; i += 256) smem[i] = 0.f;
  __syncthreads();
  const int*   es = esrc + b * E_;
  const int*   ed = edst + b * E_;
  const float* hb = h + (long)b * N_ * DIN_;
  for (int e = w; e < E_; e += 8) {             // one wave per edge
    int sx = es[e], d = ed[e];
    const float* hs = hb + (long)sx * DIN_;
    float*       ad = aggS + d * DIN_;
#pragma unroll
    for (int i = 0; i < 8; i++) atomAddLds(&ad[lane + i * 32], hs[lane + i * 32]);
    if (lane == 0) atomAddLds(&degS[d], 1.0f);
  }
  __syncthreads();
  _Float16* Ab = Af + (long)b * N_ * AK_;
  for (int idx = t; idx < N_ * AK_; idx += 256) {
    int n = idx >> 9, c = idx & 511;
    float v = (c < DIN_) ? hb[(long)n * DIN_ + c]
                         : aggS[n * DIN_ + (c - DIN_)] / fmaxf(degS[n], 1.0f);
    Ab[idx] = (_Float16)v;
  }
}

// ---------------- transpose+convert weights to f16 ----------------
__global__ __launch_bounds__(256) void convert_w_kernel(const float* __restrict__ Wf,
                                                        const float* __restrict__ Wp,
                                                        _Float16* __restrict__ WtF,
                                                        _Float16* __restrict__ WtP) {
  int idx = blockIdx.x * 256 + threadIdx.x;     // 0..131071
  int which = idx >> 16;
  int rem = idx & 65535;
  int n = rem >> 9, k = rem & 511;
  const float* W = which ? Wp : Wf;
  _Float16*   Wt = which ? WtP : WtF;
  Wt[(long)n * AK_ + k] = (_Float16)W[(long)k * K_ + n];
}

// ---------------- SAGE GEMM + normalize + relu (+ softmax) ----------------
// One workgroup = 16 rows x 128 cols; 8 waves, one 16x16 N-tile each; Kdim=512.
__global__ __launch_bounds__(256) void sage_gemm_kernel(const _Float16* __restrict__ Af,
                                                        const _Float16* __restrict__ Wt,
                                                        const float* __restrict__ bias,
                                                        int pool_path,
                                                        _Float16* __restrict__ featT,
                                                        _Float16* __restrict__ assign,
                                                        _Float16* __restrict__ assignT) {
  __shared__ _Float16 As[16][520];              // padded row stride (1040B, 16B-aligned)
  __shared__ float Zs[16][128];
  __shared__ float red[16][16];
  __shared__ float rowv[16];
  int t = threadIdx.x;
  int w = t >> 5, lane = t & 31;
  int m0 = blockIdx.x * 16;

  // stage 16x512 A-rows into LDS (32 halfs per thread)
  {
    int hidx = t * 32;                          // row constant across the 4 chunks
    int r = hidx >> 9, c = hidx & 511;
    const _Float16* g = &Af[((long)(m0 + r)) * AK_ + c];
    _Float16*       l = &As[r][c];
#ifdef USE_ASYNC_LDS
    // ASYNCcnt path (CDNA5 ISA): per-lane 16B global->LDS DMA.
    // VDST = LDS byte offset (low 32 bits of generic LDS pointer), VADDR = 64-bit global.
    unsigned laddr = (unsigned)(uintptr_t)l;
#pragma unroll
    for (int i = 0; i < 4; i++) {
      asm volatile("global_load_async_to_lds_b128 %0, %1, off"
                   :: "v"(laddr + i * 16),
                      "v"((unsigned long long)(uintptr_t)(g + i * 8))
                   : "memory");
    }
    asm volatile("s_wait_asynccnt 0x0" ::: "memory");
#else
#pragma unroll
    for (int i = 0; i < 4; i++)
      *(v8h*)(l + i * 8) = *(const v8h*)(g + i * 8);
#endif
  }
  __syncthreads();

  v8f acc = {0.f, 0.f, 0.f, 0.f, 0.f, 0.f, 0.f, 0.f};
  const _Float16* wbase = Wt + (long)(w * 16) * AK_;
#pragma unroll 4
  for (int kc = 0; kc < 16; kc++) {
    int k0 = kc * 32;
    v16h a = load_a_frag(&As[0][k0], 520, lane);
    v16h b = load_b_frag(wbase + k0, AK_, lane);
    acc = wmma_f16(a, b, acc);
  }

  // spill tile to LDS with bias (D layout: vgpr r -> M=r / r+8 by lane-half, N=lane&15)
  float bv = bias[w * 16 + (lane & 15)];
  union { v8f v; float f[8]; } u; u.v = acc;
  int mtop = (lane & 16) ? 8 : 0;
  int ncol = w * 16 + (lane & 15);
#pragma unroll
  for (int r = 0; r < 8; r++) Zs[r + mtop][ncol] = u.f[r] + bv;
  __syncthreads();

  // per-row L2 normalize + relu (16 threads per row, 8 cols each)
  int rr = t >> 4, seg = t & 15, c0 = seg * 8;
  float ssq = 0.f;
#pragma unroll
  for (int j = 0; j < 8; j++) { float z = Zs[rr][c0 + j]; ssq += z * z; }
  red[rr][seg] = ssq;
  __syncthreads();
  if (seg == 0) {
    float s = 0.f;
#pragma unroll
    for (int j = 0; j < 16; j++) s += red[rr][j];
    rowv[rr] = 1.0f / fmaxf(sqrtf(s), 1e-12f);
  }
  __syncthreads();
  float inv = rowv[rr];
  float z[8];
#pragma unroll
  for (int j = 0; j < 8; j++) z[j] = fmaxf(Zs[rr][c0 + j] * inv, 0.f);

  int gr = m0 + rr;                 // global node row (stripe never crosses a graph)
  int b = gr >> 8, nn = gr & 255;
  if (!pool_path) {
#pragma unroll
    for (int j = 0; j < 8; j++)
      featT[((long)(b * K_ + c0 + j)) * N_ + nn] = (_Float16)z[j];
  } else {
    // softmax over 128 cols
    float mx = z[0];
#pragma unroll
    for (int j = 1; j < 8; j++) mx = fmaxf(mx, z[j]);
    red[rr][seg] = mx;
    __syncthreads();
    if (seg == 0) {
      float m2 = red[rr][0];
#pragma unroll
      for (int j = 1; j < 16; j++) m2 = fmaxf(m2, red[rr][j]);
      rowv[rr] = m2;
    }
    __syncthreads();
    float rm = rowv[rr];
    float ez[8], es = 0.f;
#pragma unroll
    for (int j = 0; j < 8; j++) { ez[j] = __expf(z[j] - rm); es += ez[j]; }
    red[rr][seg] = es;
    __syncthreads();
    if (seg == 0) {
      float s = 0.f;
#pragma unroll
      for (int j = 0; j < 16; j++) s += red[rr][j];
      rowv[rr] = 1.0f / s;
    }
    __syncthreads();
    float is = rowv[rr];
#pragma unroll
    for (int j = 0; j < 8; j++) {
      _Float16 av = (_Float16)(ez[j] * is);
      assign[(long)gr * K_ + c0 + j] = av;
      assignT[((long)(b * K_ + c0 + j)) * N_ + nn] = av;
    }
  }
}

// ---------------- h_pool = assign^T @ feat, per graph ----------------
__global__ __launch_bounds__(256) void hpool_kernel(const _Float16* __restrict__ assignT,
                                                    const _Float16* __restrict__ featT,
                                                    float* __restrict__ outH) {
  int b = blockIdx.x;
  int w = threadIdx.x >> 5, lane = threadIdx.x & 31;
  const _Float16* At = assignT + (long)b * K_ * N_ + (long)(w * 16) * N_;
  const _Float16* Bt = featT   + (long)b * K_ * N_;
  v8f acc[8] = {};
  for (int kc = 0; kc < 8; kc++) {              // Kdim = 256 nodes
    int k0 = kc * 32;
    v16h a = load_a_frag(At + k0, N_, lane);
#pragma unroll
    for (int j = 0; j < 8; j++) {
      v16h bb = load_b_frag(Bt + (long)(j * 16) * N_ + k0, N_, lane);
      acc[j] = wmma_f16(a, bb, acc[j]);
    }
  }
  int mtop = (lane & 16) ? 8 : 0, nl = lane & 15;
#pragma unroll
  for (int j = 0; j < 8; j++) {
    union { v8f v; float f[8]; } u; u.v = acc[j];
#pragma unroll
    for (int r = 0; r < 8; r++) {
      int m = w * 16 + r + mtop;
      outH[((long)(b * K_ + m)) * K_ + j * 16 + nl] = u.f[r];
    }
  }
}

// ---------------- blocks = S_dst^T @ S_src over edges, per graph ----------------
// grid: 64 graphs x 4 edge-splits; double-buffered transposed LDS tiles,
// one barrier per 32-edge chunk, next gather overlapped with WMMA.
struct EdgeRegs { v8h d0, d1, s0, s1; };

__device__ __forceinline__ void gather_rows(const _Float16* __restrict__ assign,
                                            const int* __restrict__ es,
                                            const int* __restrict__ ed,
                                            int e, int c0, int bN, EdgeRegs& r) {
  int d = ed[e], sx = es[e];
  const _Float16* rd = assign + ((long)(bN + d )) * K_ + c0;
  const _Float16* rs = assign + ((long)(bN + sx)) * K_ + c0;
  r.d0 = *(const v8h*)rd;  r.d1 = *(const v8h*)(rd + 8);
  r.s0 = *(const v8h*)rs;  r.s1 = *(const v8h*)(rs + 8);
}

__device__ __forceinline__ void store_tile(_Float16 (*Sd)[32], _Float16 (*Ss)[32],
                                           const EdgeRegs& r, int el, int c0) {
  union { v8h v; _Float16 h[8]; } a, b, c, d;
  a.v = r.d0; b.v = r.d1; c.v = r.s0; d.v = r.s1;
#pragma unroll
  for (int i = 0; i < 8; i++) {
    Sd[c0 + i][el] = a.h[i];  Sd[c0 + 8 + i][el] = b.h[i];
    Ss[c0 + i][el] = c.h[i];  Ss[c0 + 8 + i][el] = d.h[i];
  }
}

__global__ __launch_bounds__(256) void blocks_kernel(const _Float16* __restrict__ assign,
                                                     const int* __restrict__ esrc,
                                                     const int* __restrict__ edst,
                                                     float* __restrict__ outAdj) {
  __shared__ _Float16 SdT[2][128][32];          // [buf][cluster k][edge]
  __shared__ _Float16 SsT[2][128][32];
  int b = blockIdx.x >> 2, s = blockIdx.x & 3;
  int t = threadIdx.x, w = t >> 5, lane = t & 31;
  const int* es = esrc + b * E_;
  const int* ed = edst + b * E_;
  int e_base = s * (E_ / 4);
  const int NCH = (E_ / 4) / 32;                // 64 chunks of 32 edges
  v8f acc[8] = {};
  int el = t >> 3;                              // local edge 0..31 (8 threads per edge)
  int c0 = (t & 7) * 16;                        // 16-col slice of the 128-wide row

  EdgeRegs cur;
  gather_rows(assign, es, ed, e_base + el, c0, b * N_, cur);
  store_tile(SdT[0], SsT[0], cur, el, c0);
  __syncthreads();

  for (int ch = 0; ch < NCH; ch++) {
    int pb = ch & 1;
    EdgeRegs nxt;
    if (ch + 1 < NCH)
      gather_rows(assign, es, ed, e_base + (ch + 1) * 32 + el, c0, b * N_, nxt);
    v16h a = load_a_frag(&SdT[pb][w * 16][0], 32, lane);
#pragma unroll
    for (int j = 0; j < 8; j++) {
      v16h bb = load_b_frag(&SsT[pb][j * 16][0], 32, lane);
      acc[j] = wmma_f16(a, bb, acc[j]);
    }
    if (ch + 1 < NCH)
      store_tile(SdT[1 - pb], SsT[1 - pb], nxt, el, c0);
    __syncthreads();
  }

  int mtop = (lane & 16) ? 8 : 0, nl = lane & 15;
#pragma unroll
  for (int j = 0; j < 8; j++) {
    union { v8f v; float f[8]; } u; u.v = acc[j];
#pragma unroll
    for (int r = 0; r < 8; r++) {
      int m = w * 16 + r + mtop;
      int n = j * 16 + nl;
      atomAddDev(&outAdj[((long)(b * K_ + m)) * ADJ_ + b * K_ + n], u.f[r]);
    }
  }
}

extern "C" void kernel_launch(void* const* d_in, const int* in_sizes, int n_in,
                              void* d_out, int out_size, void* d_ws, size_t ws_size,
                              hipStream_t stream) {
  const float* h    = (const float*)d_in[0];
  const int*   esrc = (const int*)d_in[1];
  const int*   edst = (const int*)d_in[2];
  const float* Wf   = (const float*)d_in[3];
  const float* bf   = (const float*)d_in[4];
  const float* Wp   = (const float*)d_in[5];
  const float* bp   = (const float*)d_in[6];
  float* out = (float*)d_out;

  char* ws = (char*)d_ws;
  size_t off = 0;
  _Float16* Af      = (_Float16*)(ws + off); off += (size_t)ROWS_ * AK_ * 2;   // 16 MB
  _Float16* WtF     = (_Float16*)(ws + off); off += (size_t)K_ * AK_ * 2;
  _Float16* WtP     = (_Float16*)(ws + off); off += (size_t)K_ * AK_ * 2;
  _Float16* assign  = (_Float16*)(ws + off); off += (size_t)ROWS_ * K_ * 2;    // 4 MB
  _Float16* assignT = (_Float16*)(ws + off); off += (size_t)B_ * K_ * N_ * 2;
  _Float16* featT   = (_Float16*)(ws + off); off += (size_t)B_ * K_ * N_ * 2;

  // 1) zero block-diagonal adjacency output (268MB stream pass)
  zero4_kernel<<<65536, 256, 0, stream>>>((float4*)out, (long)ADJ_ * ADJ_ / 4);
  // 2) fused scatter-mean (LDS atomics) + f16 A build, one WG per graph
  scatter_build_kernel<<<B_, 256, (size_t)(N_ * DIN_ + N_) * sizeof(float), stream>>>(
      h, esrc, edst, Af);
  // 3) W^T in f16
  convert_w_kernel<<<512, 256, 0, stream>>>(Wf, Wp, WtF, WtP);
  // 4) two SAGE GEMMs (feat, pool)
  sage_gemm_kernel<<<ROWS_ / 16, 256, 0, stream>>>(Af, WtF, bf, 0, featT, assign, assignT);
  sage_gemm_kernel<<<ROWS_ / 16, 256, 0, stream>>>(Af, WtP, bp, 1, featT, assign, assignT);
  // 5) h_pool into output tail
  hpool_kernel<<<B_, 256, 0, stream>>>(assignT, featT, out + (long)ADJ_ * ADJ_);
  // 6) diagonal adjacency blocks (atomic accumulate over 4 edge splits)
  blocks_kernel<<<B_ * 4, 256, 0, stream>>>(assign, esrc, edst, out);
}
